// graphTransformer_27805618274408
// MI455X (gfx1250) — compile-verified, matrix operand order checked by reading, not run on hardware
//
#include <hip/hip_runtime.h>

#define NN 50000
#define NE 800000
#define D 128
#define NH 8
#define OD 16

typedef __attribute__((ext_vector_type(16))) __bf16 v16bf;
typedef __attribute__((ext_vector_type(8)))  float  v8f;

// ---------------- Kernel 0: convert fp32 weights -> bf16, pre-swizzled -------
// Fragment-major layout so the proj kernel loads each B fragment as one
// contiguous 32B vector per lane (2x global_load_b128, fully coalesced):
//   index = (((mat*8 + colTile)*4 + kChunk)*32 + lane)*16 + e
//   element e of (lane l) holds W[k][n] with
//   k = kChunk*32 + 8*(l>>4) + (e<8 ? e : e+8),  n = colTile*16 + (l&15)
__global__ void gt_convert_w(const float* __restrict__ Wq,
                             const float* __restrict__ Wk,
                             const float* __restrict__ Wv,
                             __bf16* __restrict__ Wbf) {
  int i = blockIdx.x * blockDim.x + threadIdx.x;       // 0 .. 49151
  int e   = i & 15;
  int l   = (i >> 4) & 31;
  int kc  = (i >> 9) & 3;
  int ct  = (i >> 11) & 7;
  int mat = i >> 14;
  const float* srcs[3] = {Wq, Wk, Wv};
  int k = kc * 32 + ((l >> 4) << 3) + ((e < 8) ? e : (e + 8));
  int n = ct * 16 + (l & 15);
  Wbf[i] = (__bf16)srcs[mat][k * D + n];
}

// ---------------- Kernel 1: Q/K/V projection via v_wmma_f32_16x16x32_bf16 -----
// One block = 16 node rows. 8 waves; wave w owns output columns [16w, 16w+16).
// K = 128 -> 4 WMMA steps per output tile, looped over the 3 weight matrices.
__global__ void gt_proj(const float* __restrict__ state,
                        const __bf16* __restrict__ Wbf,
                        const float* __restrict__ bq,
                        const float* __restrict__ bk,
                        const float* __restrict__ bv,
                        float* __restrict__ Q,
                        float* __restrict__ K,
                        float* __restrict__ V) {
  __shared__ __bf16 ldsA[16 * D];                      // 4 KB node tile (bf16)

  const int tid  = threadIdx.x;                        // 0..255
  const int row0 = blockIdx.x * 16;                    // 50000 = 3125*16 exactly

  // load + convert the 16x128 state tile (fully uniform, EXEC stays all-ones)
  for (int i = tid; i < 16 * D; i += 256) {
    int r = i >> 7, c = i & (D - 1);
    ldsA[i] = (__bf16)state[(size_t)(row0 + r) * D + c];
  }
  __syncthreads();

  const int wave = tid >> 5;
  const int lane = tid & 31;
  const int hi   = lane >> 4;                          // lane half (0/1)
  const int mn   = lane & 15;                          // A row m / B,C col n
  const int colBase = wave * 16;

  // A fragments: element e holds k = kc*32 + 8*half + (e<8 ? e : e+8).
  // The two 8-element runs are contiguous in LDS -> 2x ds_load_b128 each.
  v16bf afrag[4];
  #pragma unroll
  for (int kc = 0; kc < 4; ++kc) {
    const int kBase = kc * 32 + hi * 8;
    #pragma unroll
    for (int e = 0; e < 16; ++e) {
      int k = kBase + ((e < 8) ? e : (e + 8));
      afrag[kc][e] = ldsA[mn * D + k];
    }
  }

  const float* biases[3] = {bq, bk, bv};
  float*       outs[3]   = {Q, K, V};
  const v16bf* Wfrag = (const v16bf*)Wbf;              // 32B-aligned fragments

  #pragma unroll
  for (int mat = 0; mat < 3; ++mat) {
    v8f acc = {};
    #pragma unroll
    for (int kc = 0; kc < 4; ++kc) {
      v16bf bfrag = Wfrag[(((mat * 8 + wave) * 4) + kc) * 32 + lane];
      acc = __builtin_amdgcn_wmma_f32_16x16x32_bf16(
          /*neg_a=*/false, afrag[kc], /*neg_b=*/false, bfrag,
          /*c_mod=*/(short)0, acc, /*reuse_a=*/false, /*reuse_b=*/false);
    }
    const float b = biases[mat][colBase + mn];
    float* O = outs[mat];
    #pragma unroll
    for (int r = 0; r < 8; ++r) {                      // D-matrix: m = r + 8*half
      int m = r + hi * 8;
      O[(size_t)(row0 + m) * D + colBase + mn] = acc[r] + b;
    }
  }
}

// ---------------- Kernel 2: zero accumulators --------------------------------
__global__ void gt_zero(float* __restrict__ p, int n) {
  int i = blockIdx.x * blockDim.x + threadIdx.x;
  if (i < n) p[i] = 0.0f;
}

// ---------------- Kernel 3: per-edge attention + scatter-add -----------------
// One edge per wave32: lane owns channels [4*lane, 4*lane+4); head = lane/4.
// Dot reduced with shfl_xor(1), shfl_xor(2). Atomics land in L2 (fits 192MB).
__global__ void gt_edge(const float* __restrict__ Q,
                        const float* __restrict__ K,
                        const float* __restrict__ V,
                        const int* __restrict__ src,
                        const int* __restrict__ dst,
                        float* __restrict__ wV,      // = d_out accumulator
                        float* __restrict__ z) {
  const int lane   = threadIdx.x & 31;
  const int wid    = (blockIdx.x * blockDim.x + threadIdx.x) >> 5;
  const int nWaves = (gridDim.x * blockDim.x) >> 5;

  for (int e = wid; e < NE; e += nWaves) {
    const int s = src[e];
    const int d = dst[e];
    const float* Ks = K + (size_t)s * D;
    const float* Qd = Q + (size_t)d * D;
    const float* Vs = V + (size_t)s * D;

    __builtin_prefetch(Vs + lane * 4, 0, 0);           // global_prefetch_b8

    const float4 kv = *(const float4*)(Ks + lane * 4);
    const float4 qv = *(const float4*)(Qd + lane * 4);
    float dot = kv.x * qv.x + kv.y * qv.y + kv.z * qv.z + kv.w * qv.w;
    dot += __shfl_xor(dot, 1, 32);
    dot += __shfl_xor(dot, 2, 32);                     // 4-lane head reduction

    float sc = dot * 0.25f;                            // / sqrt(16)
    sc = fminf(5.0f, fmaxf(-5.0f, sc));
    const float w = __expf(sc);

    const float4 vv = *(const float4*)(Vs + lane * 4);
    float* wd = wV + (size_t)d * D + lane * 4;
    atomicAdd(wd + 0, vv.x * w);
    atomicAdd(wd + 1, vv.y * w);
    atomicAdd(wd + 2, vv.z * w);
    atomicAdd(wd + 3, vv.w * w);
    if ((lane & 3) == 0)
      atomicAdd(z + (size_t)d * NH + (lane >> 2), w);
  }
}

// ---------------- Kernel 4: normalize in place -------------------------------
__global__ void gt_finalize(float* __restrict__ out, const float* __restrict__ z,
                            int n) {
  int i = blockIdx.x * blockDim.x + threadIdx.x;
  if (i < n) {
    int node = i >> 7;
    int h    = (i >> 4) & 7;
    out[i] = out[i] / z[node * NH + h];
  }
}

// ---------------- launch -----------------------------------------------------
extern "C" void kernel_launch(void* const* d_in, const int* in_sizes, int n_in,
                              void* d_out, int out_size, void* d_ws, size_t ws_size,
                              hipStream_t stream) {
  const float* state = (const float*)d_in[0];
  const int*   src   = (const int*)d_in[1];
  const int*   dst   = (const int*)d_in[2];
  const float* Wq    = (const float*)d_in[3];
  const float* bq    = (const float*)d_in[4];
  const float* Wk    = (const float*)d_in[5];
  const float* bk    = (const float*)d_in[6];
  const float* Wv    = (const float*)d_in[7];
  const float* bv    = (const float*)d_in[8];
  float* out = (float*)d_out;

  // workspace: Q | K | V | z | Wbf  (~78.5 MB)
  float*  Q   = (float*)d_ws;
  float*  K   = Q + (size_t)NN * D;
  float*  V   = K + (size_t)NN * D;
  float*  z   = V + (size_t)NN * D;
  __bf16* Wbf = (__bf16*)(z + (size_t)NN * NH);

  gt_convert_w<<<(3 * D * D) / 256, 256, 0, stream>>>(Wq, Wk, Wv, Wbf);
  gt_proj<<<NN / 16, 256, 0, stream>>>(state, Wbf, bq, bk, bv, Q, K, V);
  gt_zero<<<(NN * D + 255) / 256, 256, 0, stream>>>(out, NN * D);
  gt_zero<<<(NN * NH + 255) / 256, 256, 0, stream>>>(z, NN * NH);
  gt_edge<<<12800, 256, 0, stream>>>(Q, K, V, src, dst, out, z);
  gt_finalize<<<(NN * D + 255) / 256, 256, 0, stream>>>(out, z, NN * D);
}
